// MultiHeadAttention_12575664242838
// MI455X (gfx1250) — matrix operand order repaired and checked by reading, not compiled, not run on hardware
//
#include <hip/hip_runtime.h>

// Problem constants (from reference): B=4, S=1024, E=768, H=12, D=64
#define B_ 4
#define S_ 1024
#define E_ 768
#define H_ 12
#define D_ 64
#define HD_ (H_ * D_)   // 768

typedef __attribute__((ext_vector_type(16))) __bf16       v16bf;
typedef __attribute__((ext_vector_type(8)))  float        v8f;
typedef __attribute__((ext_vector_type(4)))  unsigned int u32x4;

union Frag {
  v16bf v;
  u32x4 q[2];
  unsigned short s[16];
};

__device__ __forceinline__ unsigned short f32_to_bf16(float f) {
  union { float f; unsigned int u; } c; c.f = f;
  unsigned int u = c.u;
  return (unsigned short)((u + 0x7FFFu + ((u >> 16) & 1u)) >> 16);
}

__device__ __forceinline__ float rowmax16(float v) {
  v = fmaxf(v, __shfl_xor(v, 8, 32));
  v = fmaxf(v, __shfl_xor(v, 4, 32));
  v = fmaxf(v, __shfl_xor(v, 2, 32));
  v = fmaxf(v, __shfl_xor(v, 1, 32));
  return v;
}
__device__ __forceinline__ float rowsum16(float v) {
  v += __shfl_xor(v, 8, 32);
  v += __shfl_xor(v, 4, 32);
  v += __shfl_xor(v, 2, 32);
  v += __shfl_xor(v, 1, 32);
  return v;
}

// ---------------- conversion kernels ----------------

__global__ void cvt_bf16_kernel(const float* __restrict__ in,
                                unsigned short* __restrict__ out, int n) {
  int i = blockIdx.x * blockDim.x + threadIdx.x;
  if (i < n) out[i] = f32_to_bf16(in[i]);
}

// in: [slices, R, C] f32  ->  out: [slices, C, R] bf16
__global__ void transpose_bf16_kernel(const float* __restrict__ in,
                                      unsigned short* __restrict__ out,
                                      int R, int C, int total) {
  int i = blockIdx.x * blockDim.x + threadIdx.x;
  if (i >= total) return;
  int rc = R * C;
  int s = i / rc;
  int rem = i - s * rc;
  int r = rem / C;
  int c = rem - r * C;
  out[(size_t)s * rc + (size_t)c * R + r] = f32_to_bf16(in[i]);
}

// ---------------- QKV projection GEMM (N-blocked x4 per wave) ----------------
// grid.x = mat(0..2)*B*H + b*H + h ; grid.y = S/64 ; block = 128
// wave w computes rows [by*64 + w*16, +16) x all 64 cols (4 accumulators).
// All 5 fragment loads per k-step issue before the 4 WMMAs -> partial waits.
__global__ __launch_bounds__(128) void qkv_gemm_kernel(
    const unsigned short* __restrict__ xb,   // [B*S, E] bf16
    const unsigned short* __restrict__ WT,   // [3, H, D, E] bf16 (transposed weights)
    unsigned short* __restrict__ Qb,         // [B, H, S, D]
    unsigned short* __restrict__ Kb,         // [B, H, S, D]
    unsigned short* __restrict__ Vtb)        // [B, H, D, S]  (V transposed)
{
  const int lane = threadIdx.x & 31;
  const int wave = threadIdx.x >> 5;
  const int lm   = lane & 15;
  const int half = lane >> 4;
  const int bx   = blockIdx.x;
  const int mat  = bx / (B_ * H_);
  const int bh   = bx % (B_ * H_);
  const int b = bh / H_, h = bh % H_;
  const int m0 = blockIdx.y * 64 + wave * 16;

  const unsigned short* xrow  = xb + ((size_t)b * S_ + m0 + lm) * E_;
  const unsigned short* wbase = WT + (((size_t)mat * H_ + h) * D_ + lm) * E_;

  v8f c0 = {}, c1 = {}, c2 = {}, c3 = {};
  #pragma unroll 4
  for (int k0 = 0; k0 < E_; k0 += 32) {
    __builtin_prefetch(xrow + k0 + 128, 0, 1);
    Frag a, w0, w1, w2, w3;
    const unsigned short* wp = wbase + k0 + half * 8;
    a.q[0]  = *(const u32x4*)(xrow + k0 + half * 8);
    a.q[1]  = *(const u32x4*)(xrow + k0 + 16 + half * 8);
    w0.q[0] = *(const u32x4*)(wp);
    w0.q[1] = *(const u32x4*)(wp + 16);
    w1.q[0] = *(const u32x4*)(wp + (size_t)16 * E_);
    w1.q[1] = *(const u32x4*)(wp + (size_t)16 * E_ + 16);
    w2.q[0] = *(const u32x4*)(wp + (size_t)32 * E_);
    w2.q[1] = *(const u32x4*)(wp + (size_t)32 * E_ + 16);
    w3.q[0] = *(const u32x4*)(wp + (size_t)48 * E_);
    w3.q[1] = *(const u32x4*)(wp + (size_t)48 * E_ + 16);
    c0 = __builtin_amdgcn_wmma_f32_16x16x32_bf16(false, a.v, false, w0.v, (short)0, c0, false, false);
    c1 = __builtin_amdgcn_wmma_f32_16x16x32_bf16(false, a.v, false, w1.v, (short)0, c1, false, false);
    c2 = __builtin_amdgcn_wmma_f32_16x16x32_bf16(false, a.v, false, w2.v, (short)0, c2, false, false);
    c3 = __builtin_amdgcn_wmma_f32_16x16x32_bf16(false, a.v, false, w3.v, (short)0, c3, false, false);
  }

  #pragma unroll
  for (int n = 0; n < 4; ++n) {
    const v8f c = (n == 0) ? c0 : (n == 1) ? c1 : (n == 2) ? c2 : c3;
    const int d = n * 16 + lm;
    #pragma unroll
    for (int r = 0; r < 8; ++r) {
      const int row = m0 + r + 8 * half;
      const unsigned short val = f32_to_bf16(c[r]);
      if (mat == 0)      Qb[(((size_t)b * H_ + h) * S_ + row) * D_ + d] = val;
      else if (mat == 1) Kb[(((size_t)b * H_ + h) * S_ + row) * D_ + d] = val;
      else               Vtb[(((size_t)b * H_ + h) * D_ + d) * S_ + row] = val;
    }
  }
}

// ---------------- flash attention (causal) ----------------
// grid.x = B*H ; grid.y = S/64 ; block = 128 (4 waves, each owns a 16-row query block)
__global__ __launch_bounds__(128) void attn_kernel(
    const unsigned short* __restrict__ Qb,   // [B,H,S,D]
    const unsigned short* __restrict__ Kb,   // [B,H,S,D]
    const unsigned short* __restrict__ Vtb,  // [B,H,D,S]
    unsigned short* __restrict__ Yb)         // [B,S,H*D]
{
  __shared__ unsigned short plds_all[4 * 16 * 32];   // per-wave 16x32 P tile

  const int lane = threadIdx.x & 31;
  const int wave = threadIdx.x >> 5;
  const int lm   = lane & 15;
  const int half = lane >> 4;
  const int bh = blockIdx.x;
  const int b = bh / H_, h = bh % H_;
  const int q0 = blockIdx.y * 64 + wave * 16;

  const unsigned short* qrow  = Qb  + (((size_t)b * H_ + h) * S_ + q0 + lm) * D_;
  const unsigned short* kbase = Kb  + (((size_t)b * H_ + h) * S_) * D_;
  const unsigned short* vbase = Vtb + (((size_t)b * H_ + h) * D_) * S_;
  unsigned short* plds = plds_all + wave * (16 * 32);

  // Q A-fragments (held for the whole row block): d 0..31 and 32..63
  Frag aq0, aq1;
  aq0.q[0] = *(const u32x4*)(qrow + half * 8);
  aq0.q[1] = *(const u32x4*)(qrow + 16 + half * 8);
  aq1.q[0] = *(const u32x4*)(qrow + 32 + half * 8);
  aq1.q[1] = *(const u32x4*)(qrow + 48 + half * 8);

  float mrun[8], lrun[8];
  #pragma unroll
  for (int r = 0; r < 8; ++r) { mrun[r] = -1e30f; lrun[r] = 0.0f; }
  v8f acc0 = {}, acc1 = {}, acc2 = {}, acc3 = {};

  const float scale = 0.125f;  // 1/sqrt(D)

  for (int t0 = 0; t0 < q0 + 16; t0 += 32) {
    // ---- scores: S[16x32] = Q[16x64] @ K_blk^T ----
    v8f s0 = {}, s1 = {};
    {
      const unsigned short* krow0 = kbase + (size_t)(t0 + lm) * D_ + half * 8;
      const unsigned short* krow1 = kbase + (size_t)(t0 + 16 + lm) * D_ + half * 8;
      Frag bk00, bk01, bk10, bk11;
      bk00.q[0] = *(const u32x4*)(krow0);
      bk00.q[1] = *(const u32x4*)(krow0 + 16);
      bk01.q[0] = *(const u32x4*)(krow0 + 32);
      bk01.q[1] = *(const u32x4*)(krow0 + 48);
      bk10.q[0] = *(const u32x4*)(krow1);
      bk10.q[1] = *(const u32x4*)(krow1 + 16);
      bk11.q[0] = *(const u32x4*)(krow1 + 32);
      bk11.q[1] = *(const u32x4*)(krow1 + 48);
      s0 = __builtin_amdgcn_wmma_f32_16x16x32_bf16(false, aq0.v, false, bk00.v, (short)0, s0, false, false);
      s0 = __builtin_amdgcn_wmma_f32_16x16x32_bf16(false, aq1.v, false, bk01.v, (short)0, s0, false, false);
      s1 = __builtin_amdgcn_wmma_f32_16x16x32_bf16(false, aq0.v, false, bk10.v, (short)0, s1, false, false);
      s1 = __builtin_amdgcn_wmma_f32_16x16x32_bf16(false, aq1.v, false, bk11.v, (short)0, s1, false, false);
    }
    #pragma unroll
    for (int r = 0; r < 8; ++r) { s0[r] *= scale; s1[r] *= scale; }

    // ---- causal mask (only blocks touching the diagonal) ----
    if (t0 + 31 > q0) {
      const int col0 = t0 + lm, col1 = t0 + 16 + lm;
      #pragma unroll
      for (int r = 0; r < 8; ++r) {
        const int row = q0 + r + 8 * half;
        if (col0 > row) s0[r] = -1e30f;
        if (col1 > row) s1[r] = -1e30f;
      }
    }

    // ---- online softmax update ----
    float nm[8], sc[8];
    #pragma unroll
    for (int r = 0; r < 8; ++r) {
      float v = rowmax16(fmaxf(s0[r], s1[r]));
      nm[r] = fmaxf(mrun[r], v);
      sc[r] = __expf(mrun[r] - nm[r]);
      mrun[r] = nm[r];
    }
    #pragma unroll
    for (int r = 0; r < 8; ++r) {
      s0[r] = __expf(s0[r] - nm[r]);
      s1[r] = __expf(s1[r] - nm[r]);
      lrun[r] = lrun[r] * sc[r] + rowsum16(s0[r] + s1[r]);
      acc0[r] *= sc[r]; acc1[r] *= sc[r]; acc2[r] *= sc[r]; acc3[r] *= sc[r];
    }

    // ---- C-layout P -> LDS -> A-layout fragment ----
    #pragma unroll
    for (int r = 0; r < 8; ++r) {
      const int row = r + 8 * half;
      plds[row * 32 + lm]      = f32_to_bf16(s0[r]);
      plds[row * 32 + 16 + lm] = f32_to_bf16(s1[r]);
    }
    asm volatile("" ::: "memory");
    asm volatile("s_wait_dscnt 0x0" ::: "memory");   // all lanes' LDS writes visible
    Frag ap;
    ap.q[0] = *(const u32x4*)(plds + lm * 32 + half * 8);
    ap.q[1] = *(const u32x4*)(plds + lm * 32 + 16 + half * 8);

    // ---- out += P[16x32] @ V_blk[32x64] ----
    {
      const unsigned short* vrow = vbase + (size_t)lm * S_ + t0 + half * 8;
      Frag bv0, bv1, bv2, bv3;
      bv0.q[0] = *(const u32x4*)(vrow);
      bv0.q[1] = *(const u32x4*)(vrow + 16);
      bv1.q[0] = *(const u32x4*)(vrow + (size_t)16 * S_);
      bv1.q[1] = *(const u32x4*)(vrow + (size_t)16 * S_ + 16);
      bv2.q[0] = *(const u32x4*)(vrow + (size_t)32 * S_);
      bv2.q[1] = *(const u32x4*)(vrow + (size_t)32 * S_ + 16);
      bv3.q[0] = *(const u32x4*)(vrow + (size_t)48 * S_);
      bv3.q[1] = *(const u32x4*)(vrow + (size_t)48 * S_ + 16);
      acc0 = __builtin_amdgcn_wmma_f32_16x16x32_bf16(false, ap.v, false, bv0.v, (short)0, acc0, false, false);
      acc1 = __builtin_amdgcn_wmma_f32_16x16x32_bf16(false, ap.v, false, bv1.v, (short)0, acc1, false, false);
      acc2 = __builtin_amdgcn_wmma_f32_16x16x32_bf16(false, ap.v, false, bv2.v, (short)0, acc2, false, false);
      acc3 = __builtin_amdgcn_wmma_f32_16x16x32_bf16(false, ap.v, false, bv3.v, (short)0, acc3, false, false);
    }
  }

  // ---- normalize and emit Y[b, s, h*64 + d] bf16 ----
  #pragma unroll
  for (int r = 0; r < 8; ++r) {
    const int row = q0 + r + 8 * half;
    const float inv = 1.0f / lrun[r];
    unsigned short* yp = Yb + ((size_t)b * S_ + row) * HD_ + h * D_;
    yp[lm]      = f32_to_bf16(acc0[r] * inv);
    yp[16 + lm] = f32_to_bf16(acc1[r] * inv);
    yp[32 + lm] = f32_to_bf16(acc2[r] * inv);
    yp[48 + lm] = f32_to_bf16(acc3[r] * inv);
  }
}

// ---------------- output projection GEMM + bias (N-blocked x4 per wave) ----------------
// grid.x = (B*S)/64 ; grid.y = E/64 ; block = 128
__global__ __launch_bounds__(128) void out_proj_kernel(
    const unsigned short* __restrict__ Yb,   // [B*S, HD]
    const unsigned short* __restrict__ WpT,  // [E, HD]  (Wp transposed)
    const float* __restrict__ bp,            // [E]
    float* __restrict__ out)                 // [B*S, E] f32
{
  const int lane = threadIdx.x & 31;
  const int wave = threadIdx.x >> 5;
  const int lm   = lane & 15;
  const int half = lane >> 4;
  const int m0   = blockIdx.x * 64 + wave * 16;
  const int col0 = blockIdx.y * 64;

  const unsigned short* yrow  = Yb  + (size_t)(m0 + lm) * HD_;
  const unsigned short* wbase = WpT + (size_t)(col0 + lm) * HD_;

  v8f c0 = {}, c1 = {}, c2 = {}, c3 = {};
  #pragma unroll 4
  for (int k0 = 0; k0 < HD_; k0 += 32) {
    __builtin_prefetch(yrow + k0 + 128, 0, 1);
    Frag a, w0, w1, w2, w3;
    const unsigned short* wp = wbase + k0 + half * 8;
    a.q[0]  = *(const u32x4*)(yrow + k0 + half * 8);
    a.q[1]  = *(const u32x4*)(yrow + k0 + 16 + half * 8);
    w0.q[0] = *(const u32x4*)(wp);
    w0.q[1] = *(const u32x4*)(wp + 16);
    w1.q[0] = *(const u32x4*)(wp + (size_t)16 * HD_);
    w1.q[1] = *(const u32x4*)(wp + (size_t)16 * HD_ + 16);
    w2.q[0] = *(const u32x4*)(wp + (size_t)32 * HD_);
    w2.q[1] = *(const u32x4*)(wp + (size_t)32 * HD_ + 16);
    w3.q[0] = *(const u32x4*)(wp + (size_t)48 * HD_);
    w3.q[1] = *(const u32x4*)(wp + (size_t)48 * HD_ + 16);
    c0 = __builtin_amdgcn_wmma_f32_16x16x32_bf16(false, a.v, false, w0.v, (short)0, c0, false, false);
    c1 = __builtin_amdgcn_wmma_f32_16x16x32_bf16(false, a.v, false, w1.v, (short)0, c1, false, false);
    c2 = __builtin_amdgcn_wmma_f32_16x16x32_bf16(false, a.v, false, w2.v, (short)0, c2, false, false);
    c3 = __builtin_amdgcn_wmma_f32_16x16x32_bf16(false, a.v, false, w3.v, (short)0, c3, false, false);
  }

  #pragma unroll
  for (int n = 0; n < 4; ++n) {
    const v8f c = (n == 0) ? c0 : (n == 1) ? c1 : (n == 2) ? c2 : c3;
    const int col = col0 + n * 16 + lm;
    const float bias = bp[col];
    #pragma unroll
    for (int r = 0; r < 8; ++r) {
      const int row = m0 + r + 8 * half;
      out[(size_t)row * E_ + col] = c[r] + bias;
    }
  }
}

// ---------------- host launch ----------------

extern "C" void kernel_launch(void* const* d_in, const int* in_sizes, int n_in,
                              void* d_out, int out_size, void* d_ws, size_t ws_size,
                              hipStream_t stream) {
  const float* x  = (const float*)d_in[0];   // [B,S,E]
  const float* Wq = (const float*)d_in[1];   // [H,E,D]
  const float* Wk = (const float*)d_in[2];   // [H,E,D]
  const float* Wv = (const float*)d_in[3];   // [H,E,D]
  const float* Wp = (const float*)d_in[4];   // [HD,E]
  const float* bp = (const float*)d_in[5];   // [E]
  float* out = (float*)d_out;                // [B,S,E] f32

  // workspace layout (bf16 = unsigned short), every section 16B-aligned
  unsigned short* ws = (unsigned short*)d_ws;
  size_t o = 0;
  unsigned short* xb   = ws + o; o += (size_t)B_ * S_ * E_;        // x bf16
  unsigned short* WT   = ws + o; o += (size_t)3 * H_ * D_ * E_;    // Wq/Wk/Wv transposed
  unsigned short* WpT  = ws + o; o += (size_t)E_ * HD_;            // Wp transposed
  unsigned short* Qb   = ws + o; o += (size_t)B_ * H_ * S_ * D_;
  unsigned short* Kb   = ws + o; o += (size_t)B_ * H_ * S_ * D_;
  unsigned short* Vtb  = ws + o; o += (size_t)B_ * H_ * D_ * S_;
  unsigned short* Yb   = ws + o; o += (size_t)B_ * S_ * HD_;
  (void)ws_size; (void)in_sizes; (void)n_in; (void)out_size;

  // 1) convert x to bf16
  {
    int n = B_ * S_ * E_;
    cvt_bf16_kernel<<<(n + 255) / 256, 256, 0, stream>>>(x, xb, n);
  }
  // 2) transpose-convert weights
  {
    int total = H_ * E_ * D_;
    int g = (total + 255) / 256;
    transpose_bf16_kernel<<<g, 256, 0, stream>>>(Wq, WT + 0 * (size_t)H_ * D_ * E_, E_, D_, total);
    transpose_bf16_kernel<<<g, 256, 0, stream>>>(Wk, WT + 1 * (size_t)H_ * D_ * E_, E_, D_, total);
    transpose_bf16_kernel<<<g, 256, 0, stream>>>(Wv, WT + 2 * (size_t)H_ * D_ * E_, E_, D_, total);
    int totalP = HD_ * E_;
    transpose_bf16_kernel<<<(totalP + 255) / 256, 256, 0, stream>>>(Wp, WpT, HD_, E_, totalP);
  }
  // 3) QKV projections (WMMA, N-blocked)
  {
    dim3 grid(3 * B_ * H_, S_ / 64);
    qkv_gemm_kernel<<<grid, 128, 0, stream>>>(xb, WT, Qb, Kb, Vtb);
  }
  // 4) causal flash attention (WMMA)
  {
    dim3 grid(B_ * H_, S_ / 64);
    attn_kernel<<<grid, 128, 0, stream>>>(Qb, Kb, Vtb, Yb);
  }
  // 5) output projection + bias (WMMA, N-blocked)
  {
    dim3 grid((B_ * S_) / 64, E_ / 64);
    out_proj_kernel<<<grid, 128, 0, stream>>>(Yb, WpT, bp, out);
  }
}
